// BaseAttentionEncoder_44633300140321
// MI455X (gfx1250) — compile-verified, matrix operand order checked by reading, not run on hardware
//
#include <hip/hip_runtime.h>
#include <hip/hip_bf16.h>

// ---------------------------------------------------------------------------
// BaseAttentionEncoder on gfx1250 (MI455X): bf16 WMMA GEMMs, f32 accumulate.
//   Q = x Wq^T + bq        (f32 -> S_p[:, 0:L, :], bf16 -> ws)
//   K = x Wk^T + bk        (bf16 -> ws)
//   V = x Wv^T + bv        (bf16, TRANSPOSED [B][D][L] -> ws)
//   E = Q K^T / 16         (f32 -> d_out A region)
//   A = softmax(E, axis=q) (in place, column softmax)
//   C = A V                (f32 -> S_p[:, L:2L, :])
// ---------------------------------------------------------------------------

constexpr int Bb = 8;
constexpr int Ls = 2048;
constexpr int Dd = 256;

typedef __attribute__((ext_vector_type(16))) __bf16 v16bf;
typedef __attribute__((ext_vector_type(8)))  __bf16 bf16x8;
typedef __attribute__((ext_vector_type(8)))  float  v8f;

__device__ __forceinline__ v8f wmma_bf16(v16bf a, v16bf b, v8f c) {
    return __builtin_amdgcn_wmma_f32_16x16x32_bf16(
        /*neg_a=*/false, a, /*neg_b=*/false, b,
        /*c_mod=*/(short)0, c, /*reuse_a=*/false, /*reuse_b=*/false);
}

// 16x32 bf16 fragment (A or B side, symmetric layout), row already resolved.
// Lane layout: VGPR0..3 <- K = half*8 + 0..7 ; VGPR4..7 <- K = 16 + half*8 + 0..7
__device__ __forceinline__ v16bf load_frag_bf16(const __bf16* __restrict__ row, int half) {
    bf16x8 lo = *(const bf16x8*)(row + half * 8);
    bf16x8 hi = *(const bf16x8*)(row + 16 + half * 8);
    return __builtin_shufflevector(lo, hi, 0,1,2,3,4,5,6,7,8,9,10,11,12,13,14,15);
}

// Same fragment but sourced from f32 memory, converted to bf16 in registers.
__device__ __forceinline__ v16bf load_frag_f32(const float* __restrict__ row, int half) {
    const float* p0 = row + half * 8;
    const float* p1 = row + 16 + half * 8;
    bf16x8 lo, hi;
#pragma unroll
    for (int i = 0; i < 8; ++i) { lo[i] = (__bf16)p0[i]; hi[i] = (__bf16)p1[i]; }
    return __builtin_shufflevector(lo, hi, 0,1,2,3,4,5,6,7,8,9,10,11,12,13,14,15);
}

// ---------------------------------------------------------------------------
// Kernel 1: QKV projections.  grid = (2048, 3), 256 threads = 8 waves.
// Wave tile decomposition: b(8) x lt(128) x ot(16) -> 16384 waves per proj.
// proj 0/1 (Q/K): D-tile = [o x l] (A = W rows, B = x rows) -> lane holds
//                 8 consecutive o for fixed l -> contiguous [l][o] stores.
// proj 2   (V):   D-tile = [l x d] (A = x rows, B = W rows) -> lane holds
//                 8 consecutive l for fixed d -> contiguous V^T [d][l] store.
// ---------------------------------------------------------------------------
__global__ __launch_bounds__(256) void qkv_proj_kernel(
    const float* __restrict__ x,
    const float* __restrict__ Wq, const float* __restrict__ bq,
    const float* __restrict__ Wk, const float* __restrict__ bk,
    const float* __restrict__ Wv, const float* __restrict__ bv,
    float* __restrict__ Sp,
    __bf16* __restrict__ Qb, __bf16* __restrict__ Kb, __bf16* __restrict__ Vt)
{
    const int proj = blockIdx.y;                       // 0=Q 1=K 2=V
    const int wid  = blockIdx.x * 8 + (threadIdx.x >> 5);
    const int lane = threadIdx.x & 31;
    const int idx  = lane & 15;
    const int half = lane >> 4;

    int t = wid;
    const int ot = t & 15;  t >>= 4;
    const int lt = t & 127; t >>= 7;
    const int b  = t;
    const int l0 = lt * 16, o0 = ot * 16;

    const float* W    = (proj == 0) ? Wq : (proj == 1) ? Wk : Wv;
    const float* bias = (proj == 0) ? bq : (proj == 1) ? bk : bv;
    const float* xb   = x + (size_t)b * Ls * Dd;

    v8f acc = {};
    if (proj < 2) {
        const float* wrow = W  + (size_t)(o0 + idx) * Dd;   // A: W row o
        const float* xrow = xb + (size_t)(l0 + idx) * Dd;   // B: x row l
#pragma unroll
        for (int kb = 0; kb < Dd; kb += 32) {
            v16bf af = load_frag_f32(wrow + kb, half);
            v16bf bf = load_frag_f32(xrow + kb, half);
            acc = wmma_bf16(af, bf, acc);
        }
        const int ob   = o0 + half * 8;     // 8 consecutive o per lane
        const int lcol = l0 + idx;          // fixed l per lane
#pragma unroll
        for (int r = 0; r < 8; ++r) acc[r] += bias[ob + r];

        bf16x8 pk;
#pragma unroll
        for (int r = 0; r < 8; ++r) pk[r] = (__bf16)acc[r];

        if (proj == 0) {
            *(v8f*)(Sp + ((size_t)b * 2 * Ls + lcol) * Dd + ob) = acc;      // f32 Q into S_p
            *(bf16x8*)(Qb + ((size_t)b * Ls + lcol) * Dd + ob) = pk;        // bf16 Q
        } else {
            *(bf16x8*)(Kb + ((size_t)b * Ls + lcol) * Dd + ob) = pk;        // bf16 K
        }
    } else {
        const float* xrow = xb + (size_t)(l0 + idx) * Dd;   // A: x row l
        const float* wrow = W  + (size_t)(o0 + idx) * Dd;   // B: Wv row d
#pragma unroll
        for (int kb = 0; kb < Dd; kb += 32) {
            v16bf af = load_frag_f32(xrow + kb, half);
            v16bf bf = load_frag_f32(wrow + kb, half);
            acc = wmma_bf16(af, bf, acc);
        }
        const float bvs = bias[o0 + idx];   // fixed d per lane
        bf16x8 pk;
#pragma unroll
        for (int r = 0; r < 8; ++r) pk[r] = (__bf16)(acc[r] + bvs);
        // V^T store: [b][d][l], lane holds 8 consecutive l
        *(bf16x8*)(Vt + ((size_t)b * Dd + o0 + idx) * Ls + l0 + half * 8) = pk;
    }
}

// ---------------------------------------------------------------------------
// Kernel 2: E = Q K^T / sqrt(D).  NT GEMM over d (both operands d-contiguous).
// 2x2 register blocking: wave computes 32(k) x 32(q): 2 K-frags + 2 Q-frags
// feed 4 WMMAs per K-step (1 wmma per fragment load).
// D-tile = [k x q] -> lane holds 8 consecutive k for fixed q -> contiguous
// E[b][q][k] v8f stores.  waves: b(8) x kp(64) x qp(64) -> 4096 blocks.
// ---------------------------------------------------------------------------
__global__ __launch_bounds__(256) void attn_scores_kernel(
    const __bf16* __restrict__ Qb, const __bf16* __restrict__ Kb,
    float* __restrict__ E)
{
    const int wid  = blockIdx.x * 8 + (threadIdx.x >> 5);
    const int lane = threadIdx.x & 31;
    const int idx  = lane & 15;
    const int half = lane >> 4;

    int t = wid;
    const int qp = t & 63; t >>= 6;
    const int kp = t & 63; t >>= 6;
    const int b  = t;
    const int k0 = kp * 32, q0 = qp * 32;

    const __bf16* Krow0 = Kb + ((size_t)b * Ls + k0 + idx) * Dd;
    const __bf16* Krow1 = Krow0 + (size_t)16 * Dd;
    const __bf16* Qrow0 = Qb + ((size_t)b * Ls + q0 + idx) * Dd;
    const __bf16* Qrow1 = Qrow0 + (size_t)16 * Dd;

    v8f acc00 = {}, acc01 = {}, acc10 = {}, acc11 = {};   // [ktile][qtile]
#pragma unroll
    for (int kb = 0; kb < Dd; kb += 32) {
        v16bf a0 = load_frag_bf16(Krow0 + kb, half);
        v16bf a1 = load_frag_bf16(Krow1 + kb, half);
        v16bf b0 = load_frag_bf16(Qrow0 + kb, half);
        v16bf b1 = load_frag_bf16(Qrow1 + kb, half);
        acc00 = wmma_bf16(a0, b0, acc00);
        acc01 = wmma_bf16(a0, b1, acc01);
        acc10 = wmma_bf16(a1, b0, acc10);
        acc11 = wmma_bf16(a1, b1, acc11);
    }
    const float scale = 0.0625f;   // 1/sqrt(256)
    acc00 = acc00 * scale; acc01 = acc01 * scale;
    acc10 = acc10 * scale; acc11 = acc11 * scale;
    // lane: fixed q = q0(+16)+idx, 8 consecutive k = k0(+16) + half*8 + r
    float* e00 = E + ((size_t)b * Ls + q0 + idx)      * Ls + k0 + half * 8;
    float* e01 = E + ((size_t)b * Ls + q0 + 16 + idx) * Ls + k0 + half * 8;
    *(v8f*)(e00)      = acc00;
    *(v8f*)(e01)      = acc01;
    *(v8f*)(e00 + 16) = acc10;
    *(v8f*)(e01 + 16) = acc11;
}

// ---------------------------------------------------------------------------
// Kernel 3: column softmax (axis = q).  One thread per (b, k) column; strided
// over q (8 KB stride) but coalesced across the wave. Manual prefetch since
// the strided walk defeats hardware streaming. In place in d_out's A region.
// ---------------------------------------------------------------------------
__global__ __launch_bounds__(256) void softmax_cols_kernel(float* __restrict__ A)
{
    const int col = blockIdx.x * blockDim.x + threadIdx.x;   // 0 .. B*L-1
    const int b = col / Ls;
    const int k = col % Ls;
    float* base = A + (size_t)b * Ls * Ls + k;

    float m = -3.402823466e+38f;
    for (int q = 0; q < Ls; ++q) {
        __builtin_prefetch(base + (size_t)(q + 16) * Ls, 0, 0);
        m = fmaxf(m, base[(size_t)q * Ls]);
    }
    float s = 0.0f;
    for (int q = 0; q < Ls; ++q) s += __expf(base[(size_t)q * Ls] - m);
    const float inv = 1.0f / s;
    for (int q = 0; q < Ls; ++q) {
        float v = base[(size_t)q * Ls];
        base[(size_t)q * Ls] = __expf(v - m) * inv;
    }
}

// ---------------------------------------------------------------------------
// Kernel 4: C = A V.  NT GEMM over k (2048) using V^T (k-contiguous bf16) and
// A rows (k-contiguous f32 -> bf16 in registers).  Register-blocked over d:
// each wave computes 4 d-tiles (64 d) x 16 q, so one f32 A-fragment feeds
// 4 WMMAs -> A matrix traffic cut 4x vs. one-tile waves.
// D-tile = [d x q] -> lane holds 8 consecutive d for fixed q -> contiguous
// S_p[b][L+q][d] v8f stores.  waves: b(8) x qt(128) x dp(4) -> 512 blocks.
// ---------------------------------------------------------------------------
__global__ __launch_bounds__(256) void attn_out_kernel(
    const float* __restrict__ A, const __bf16* __restrict__ Vt,
    float* __restrict__ Sp)
{
    const int wid  = blockIdx.x * 8 + (threadIdx.x >> 5);
    const int lane = threadIdx.x & 31;
    const int idx  = lane & 15;
    const int half = lane >> 4;

    int t = wid;
    const int dp = t & 3;   t >>= 2;
    const int qt = t & 127; t >>= 7;
    const int b  = t;
    const int d0 = dp * 64, q0 = qt * 16;

    const __bf16* Vrow0 = Vt + ((size_t)b * Dd + d0 + idx) * Ls;  // A-frag rows d
    const float*  Arow  = A  + ((size_t)b * Ls + q0 + idx) * Ls;  // B-frag rows q

    v8f acc[4] = {{}, {}, {}, {}};
#pragma unroll 4
    for (int kb = 0; kb < Ls; kb += 32) {
        v16bf bf = load_frag_f32(Arow + kb, half);   // shared across 4 d-tiles
#pragma unroll
        for (int j = 0; j < 4; ++j) {
            v16bf af = load_frag_bf16(Vrow0 + (size_t)(j * 16) * Ls + kb, half);
            acc[j] = wmma_bf16(af, bf, acc[j]);
        }
    }
    // lane: fixed q = q0+idx, 8 consecutive d = d0 + j*16 + half*8 + r
    float* dst = Sp + ((size_t)b * 2 * Ls + Ls + q0 + idx) * Dd + d0 + half * 8;
#pragma unroll
    for (int j = 0; j < 4; ++j) *(v8f*)(dst + j * 16) = acc[j];
}

// ---------------------------------------------------------------------------
extern "C" void kernel_launch(void* const* d_in, const int* in_sizes, int n_in,
                              void* d_out, int out_size, void* d_ws, size_t ws_size,
                              hipStream_t stream) {
    const float* x  = (const float*)d_in[0];
    const float* Wq = (const float*)d_in[1];
    const float* bq = (const float*)d_in[2];
    const float* Wk = (const float*)d_in[3];
    const float* bk = (const float*)d_in[4];
    const float* Wv = (const float*)d_in[5];
    const float* bv = (const float*)d_in[6];

    float* out = (float*)d_out;
    float* Sp  = out;                                   // [B][2L][D]
    float* Amt = out + (size_t)Bb * 2 * Ls * Dd;        // [B][L][L]

    // Workspace: bf16 Q, K (row-major [B][L][D]) and V^T ([B][D][L]) = 25.2 MB
    __bf16* Qb = (__bf16*)d_ws;
    __bf16* Kb = Qb + (size_t)Bb * Ls * Dd;
    __bf16* Vt = Kb + (size_t)Bb * Ls * Dd;

    dim3 blk(256);
    qkv_proj_kernel<<<dim3(2048, 3), blk, 0, stream>>>(x, Wq, bq, Wk, bk, Wv, bv,
                                                       Sp, Qb, Kb, Vt);
    attn_scores_kernel<<<dim3(4096), blk, 0, stream>>>(Qb, Kb, Amt);
    softmax_cols_kernel<<<dim3((Bb * Ls) / 256), blk, 0, stream>>>(Amt);
    attn_out_kernel<<<dim3(512), blk, 0, stream>>>(Amt, Vt, Sp);
}
